// TransformerEncoderLayer_33921651704166
// MI455X (gfx1250) — matrix operand order repaired and checked by reading, not compiled, not run on hardware
//
#include <hip/hip_runtime.h>
#include <hip/hip_bf16.h>
#include <cstdint>
#include <cstddef>

// ---------------------------------------------------------------------------
// Transformer encoder layer (pre-LN) for MI455X / gfx1250, bf16 WMMA path.
// B=4, S=2048, D_MODEL=512, N_HEADS=8, D_HEAD=64, D_FF=2048.
// Workspace requirement: ~64 MB.
// ---------------------------------------------------------------------------

#define D_MODEL 512
#define D_FF    2048
#define N_HEADS 8
#define D_HEAD  64
#define SEQ     2048
#define BATCH   4
#define TOKENS  (BATCH * SEQ)
#define BH      (BATCH * N_HEADS)

typedef __attribute__((ext_vector_type(16))) __bf16 bf16x16;
typedef __attribute__((ext_vector_type(8)))  __bf16 bf16x8;
typedef __attribute__((ext_vector_type(8)))  float  f32x8;

static __device__ __forceinline__ bf16x16 join8(bf16x8 lo, bf16x8 hi) {
  return __builtin_shufflevector(lo, hi, 0,1,2,3,4,5,6,7,8,9,10,11,12,13,14,15);
}
static __device__ __forceinline__ bf16x8 ld8(const __bf16* p) {
  return *reinterpret_cast<const bf16x8*>(p);           // 16B aligned by construction
}
static __device__ __forceinline__ bf16x16 ld16(const __bf16* p) {
  return join8(ld8(p), ld8(p + 8));
}
static __device__ __forceinline__ f32x8 wmma_bf16(bf16x16 a, bf16x16 b, f32x8 c) {
  // D = A(16x32) * B(32x16) + C(16x16), f32 accumulate
  return __builtin_amdgcn_wmma_f32_16x16x32_bf16(false, a, false, b, (short)0, c,
                                                 false, false);
}

// ---------------------------------------------------------------------------
// Pack fp32 weight [K x N] row-major into WMMA B-operand tile layout (bf16).
// Tile (kt, nt) of 32(K) x 16(N); base = (nt*(K/32)+kt)*512 elems;
// lane = (n%16) + 16*((k%32)>=16), elem = k%16.  One-time cost, tiny.
// ---------------------------------------------------------------------------
__global__ void pack_w(const float* __restrict__ W, __bf16* __restrict__ out,
                       int K, int N) {
  int idx = blockIdx.x * 256 + threadIdx.x;
  if (idx >= K * N) return;
  int k = idx / N, n = idx % N;
  int kt = k >> 5, kk = k & 31, nt = n >> 4, nn = n & 15;
  int lane = nn + ((kk >> 4) << 4);
  int e = kk & 15;
  out[(((size_t)nt * (K >> 5) + kt) << 9) + lane * 16 + e] = (__bf16)W[idx];
}

// ---------------------------------------------------------------------------
// LayerNorm (Bessel-corrected std, /(sigma+eps)) -> bf16 row-major copy.
// One 256-thread block per row of 512.
// ---------------------------------------------------------------------------
__global__ void ln_kernel(const float* __restrict__ X, const float* __restrict__ gamma,
                          const float* __restrict__ beta, __bf16* __restrict__ out) {
  __shared__ float red[256];
  const int row = blockIdx.x, tid = threadIdx.x;
  const float* xr = X + (size_t)row * D_MODEL;
  float x0 = xr[tid], x1 = xr[tid + 256];
  red[tid] = x0 + x1;
  __syncthreads();
  for (int s = 128; s > 0; s >>= 1) {
    if (tid < s) red[tid] += red[tid + s];
    __syncthreads();
  }
  float mu = red[0] * (1.0f / D_MODEL);
  __syncthreads();
  float d0 = x0 - mu, d1 = x1 - mu;
  red[tid] = d0 * d0 + d1 * d1;
  __syncthreads();
  for (int s = 128; s > 0; s >>= 1) {
    if (tid < s) red[tid] += red[tid + s];
    __syncthreads();
  }
  float sigma = sqrtf(red[0] * (1.0f / (D_MODEL - 1)));   // ddof=1
  float inv = 1.0f / (sigma + 1e-6f);
  __bf16* orow = out + (size_t)row * D_MODEL;
  orow[tid]       = (__bf16)(gamma[tid] * d0 * inv + beta[tid]);
  orow[tid + 256] = (__bf16)(gamma[tid + 256] * d1 * inv + beta[tid + 256]);
}

// ---------------------------------------------------------------------------
// Generic GEMM: C[M,N] = A[M,K](bf16,row-major) x Bp(packed bf16) (+bias)(+relu)
// (+fp32 residual). Each wave computes a 16x64 tile (A frag reused x4).
// ---------------------------------------------------------------------------
__global__ void gemm_ws(const __bf16* __restrict__ A, const __bf16* __restrict__ Bp,
                        const float* __restrict__ bias, const float* __restrict__ res,
                        float* __restrict__ outf, __bf16* __restrict__ outb,
                        int M, int K, int N, int relu) {
  const int lane = threadIdx.x & 31, lo = lane & 15, hi = lane >> 4;
  const int wg = blockIdx.x * (blockDim.x >> 5) + (threadIdx.x >> 5);
  const int tilesN = N >> 6;
  const int mt = wg / tilesN;
  if (mt >= (M >> 4)) return;                 // wave-uniform
  const int nt0 = (wg % tilesN) << 2;         // in 16-col units
  const int m0 = mt << 4;
  const int kt32 = K >> 5;
  const __bf16* pa = A + (size_t)(m0 + lo) * K;
  f32x8 c[4] = {};
  for (int kt = 0; kt < kt32; ++kt) {
    const int k0 = kt << 5;
    bf16x16 a = ld16(pa + k0 + 16 * hi);
    const __bf16* pb = Bp + ((size_t)nt0 * kt32 + kt) * 512 + lane * 16;
#pragma unroll
    for (int i = 0; i < 4; ++i)
      c[i] = wmma_bf16(a, ld16(pb + (size_t)i * kt32 * 512), c[i]);
  }
#pragma unroll
  for (int i = 0; i < 4; ++i) {
    const int n = ((nt0 + i) << 4) + lo;
    const float bvl = bias ? bias[n] : 0.0f;
#pragma unroll
    for (int j = 0; j < 8; ++j) {
      const int m = m0 + j + 8 * hi;
      float v = c[i][j] + bvl;
      if (relu) v = v > 0.0f ? v : 0.0f;
      if (res) v += res[(size_t)m * N + n];
      if (outf) outf[(size_t)m * N + n] = v;
      if (outb) outb[(size_t)m * N + n] = (__bf16)v;
    }
  }
}

// ---------------------------------------------------------------------------
// Fused QKV GEMM: h[TOKENS,512] x {Wq,Wk,Wv}(packed) + bias, scatter epilogue:
//   Q,K -> [BH, S, 64] bf16 (row-major)     (A / B operands of Q*K^T)
//   V   -> [BH, 64, S] bf16 (transposed)    (B operand of P*V, contiguous in k)
// ---------------------------------------------------------------------------
__global__ void gemm_qkv(const __bf16* __restrict__ A,
                         const __bf16* __restrict__ Pq, const __bf16* __restrict__ Pk,
                         const __bf16* __restrict__ Pv,
                         const float* __restrict__ bq, const float* __restrict__ bk,
                         const float* __restrict__ bvp,
                         __bf16* __restrict__ Qb, __bf16* __restrict__ Kb,
                         __bf16* __restrict__ Vt) {
  const int lane = threadIdx.x & 31, lo = lane & 15, hi = lane >> 4;
  const int wg = blockIdx.x * 8 + (threadIdx.x >> 5);
  const int tilesN = (3 * D_MODEL) >> 6;      // 24
  const int mt = wg / tilesN;
  const int nt0 = (wg % tilesN) << 2;
  const int m0 = mt << 4;
  const int kt32 = D_MODEL >> 5;              // 16
  const __bf16* pa = A + (size_t)(m0 + lo) * D_MODEL;
  f32x8 c[4] = {};
  for (int kt = 0; kt < kt32; ++kt) {
    const int k0 = kt << 5;
    bf16x16 a = ld16(pa + k0 + 16 * hi);
#pragma unroll
    for (int i = 0; i < 4; ++i) {
      const int n16 = nt0 + i;                        // global 16-col tile, 0..95
      const __bf16* pk = (n16 < 32) ? Pq : (n16 < 64) ? Pk : Pv;
      const int n16l = n16 & 31;
      c[i] = wmma_bf16(a, ld16(pk + ((size_t)n16l * kt32 + kt) * 512 + lane * 16), c[i]);
    }
  }
#pragma unroll
  for (int i = 0; i < 4; ++i) {
    const int n = ((nt0 + i) << 4) + lo;      // 0..1535
    const int which = n >> 9, f = n & 511;
    const int head = f >> 6, d = f & 63;
    const float* bias = (which == 0) ? bq : (which == 1) ? bk : bvp;
    const float bvl = bias[f];
#pragma unroll
    for (int j = 0; j < 8; ++j) {
      const int t = m0 + j + 8 * hi;          // token 0..8191
      const int bb = t >> 11, s = t & 2047;
      const int bh = bb * N_HEADS + head;
      const __bf16 v = (__bf16)(c[i][j] + bvl);
      if (which == 0)      Qb[((size_t)bh * SEQ + s) * D_HEAD + d] = v;
      else if (which == 1) Kb[((size_t)bh * SEQ + s) * D_HEAD + d] = v;
      else                 Vt[((size_t)bh * D_HEAD + d) * SEQ + s] = v;
    }
  }
}

// ---------------------------------------------------------------------------
// Flash attention: grid = BH * (S/64), 128 threads = 4 waves, each wave owns a
// 16-query tile. Per 32-key step: 4 score WMMAs, mask, online softmax (shfl
// reductions over the 16-lane halves), probs C-layout -> A-layout via LDS
// (per-wave, ordered with s_wait_dscnt), 4 PV WMMAs. ctx -> [TOKENS,512] bf16.
// ---------------------------------------------------------------------------
__global__ void attn_kernel(const __bf16* __restrict__ Qb, const __bf16* __restrict__ Kb,
                            const __bf16* __restrict__ Vt,
                            const unsigned char* __restrict__ mask,  // numpy bool
                            __bf16* __restrict__ ctx) {
  __shared__ __bf16 plds[4][16][32];          // 4 KB, one 16x32 prob tile per wave
  const int lane = threadIdx.x & 31, lo = lane & 15, hi = lane >> 4;
  const int wave = threadIdx.x >> 5;
  const int qblk = blockIdx.x & 31;           // S/64 = 32
  const int bh = blockIdx.x >> 5;
  const int b = bh >> 3, head = bh & 7;
  const int q0 = qblk * 64 + wave * 16;

  // Q A-fragments for d = 0..31 and 32..63
  const __bf16* pq = Qb + ((size_t)bh * SEQ + q0 + lo) * D_HEAD;
  const bf16x16 aq0 = ld16(pq + 16 * hi);
  const bf16x16 aq1 = ld16(pq + 32 + 16 * hi);

  float rm[8], rs[8];
#pragma unroll
  for (int j = 0; j < 8; ++j) { rm[j] = -3.0e38f; rs[j] = 0.0f; }
  f32x8 o[4] = {};

  const unsigned char* mrow = mask + ((size_t)b * SEQ + q0 + 8 * hi) * SEQ;

  for (int k0 = 0; k0 < SEQ; k0 += 32) {
    // ---- scores: two 16x16 C tiles (keys k0..+15, k0+16..+31) ----
    f32x8 s[2];
#pragma unroll
    for (int ck = 0; ck < 2; ++ck) {
      const __bf16* pk = Kb + ((size_t)bh * SEQ + k0 + ck * 16 + lo) * D_HEAD + 16 * hi;
      f32x8 z = {};
      z = wmma_bf16(aq0, ld16(pk), z);        // d 0..31
      z = wmma_bf16(aq1, ld16(pk + 32), z);   // d 32..63
      s[ck] = z;
    }
    // ---- scale + mask ----
#pragma unroll
    for (int ck = 0; ck < 2; ++ck)
#pragma unroll
      for (int j = 0; j < 8; ++j) {
        const unsigned char mv = mrow[(size_t)j * SEQ + k0 + ck * 16 + lo];
        const float v = s[ck][j] * 0.125f;    // 1/sqrt(64)
        s[ck][j] = mv ? v : -1.0e9f;
      }
    // ---- online softmax ----
#pragma unroll
    for (int j = 0; j < 8; ++j) {
      float tm = fmaxf(s[0][j], s[1][j]);
#pragma unroll
      for (int d = 1; d < 16; d <<= 1) tm = fmaxf(tm, __shfl_xor(tm, d, 32));
      const float nm = fmaxf(rm[j], tm);
      const float corr = __expf(rm[j] - nm);
      const float p0 = __expf(s[0][j] - nm);
      const float p1 = __expf(s[1][j] - nm);
      s[0][j] = p0; s[1][j] = p1;
      float ps = p0 + p1;
#pragma unroll
      for (int d = 1; d < 16; d <<= 1) ps += __shfl_xor(ps, d, 32);
      rs[j] = rs[j] * corr + ps;
      rm[j] = nm;
#pragma unroll
      for (int t = 0; t < 4; ++t) o[t][j] *= corr;
    }
    // ---- probs: C layout -> A layout through per-wave LDS tile ----
#pragma unroll
    for (int ck = 0; ck < 2; ++ck)
#pragma unroll
      for (int j = 0; j < 8; ++j)
        plds[wave][j + 8 * hi][ck * 16 + lo] = (__bf16)s[ck][j];
    asm volatile("s_wait_dscnt 0" ::: "memory");   // stores visible to own wave
    const bf16x16 pa = ld16(&plds[wave][lo][16 * hi]);
    asm volatile("s_wait_dscnt 0" ::: "memory");   // loads done before next stores
    // ---- PV: ctx(16x64) += P(16x32) x V(32x64), V transposed in memory ----
#pragma unroll
    for (int t = 0; t < 4; ++t) {
      const __bf16* pv = Vt + ((size_t)bh * D_HEAD + 16 * t + lo) * SEQ + k0 + 16 * hi;
      o[t] = wmma_bf16(pa, ld16(pv), o[t]);
    }
  }
  // ---- normalize + store ctx [TOKENS, D_MODEL] bf16 ----
#pragma unroll
  for (int j = 0; j < 8; ++j) {
    const float inv = 1.0f / rs[j];
    const size_t tok = (size_t)b * SEQ + q0 + j + 8 * hi;
#pragma unroll
    for (int t = 0; t < 4; ++t)
      ctx[tok * D_MODEL + head * D_HEAD + 16 * t + lo] = (__bf16)(o[t][j] * inv);
  }
}

// ---------------------------------------------------------------------------
// Host orchestration
// ---------------------------------------------------------------------------
extern "C" void kernel_launch(void* const* d_in, const int* in_sizes, int n_in,
                              void* d_out, int out_size, void* d_ws, size_t ws_size,
                              hipStream_t stream) {
  (void)in_sizes; (void)n_in; (void)out_size; (void)ws_size;
  const float* x    = (const float*)d_in[0];
  const unsigned char* mask = (const unsigned char*)d_in[1];  // bool, 1 byte
  const float* g1   = (const float*)d_in[2];
  const float* be1  = (const float*)d_in[3];
  const float* Wq   = (const float*)d_in[4];
  const float* bq   = (const float*)d_in[5];
  const float* Wk   = (const float*)d_in[6];
  const float* bk   = (const float*)d_in[7];
  const float* Wv   = (const float*)d_in[8];
  const float* bv   = (const float*)d_in[9];
  const float* Wo   = (const float*)d_in[10];
  const float* bo   = (const float*)d_in[11];
  const float* g2   = (const float*)d_in[12];
  const float* be2  = (const float*)d_in[13];
  const float* W1   = (const float*)d_in[14];
  const float* b1   = (const float*)d_in[15];
  const float* W2   = (const float*)d_in[16];
  const float* b2   = (const float*)d_in[17];
  float* out = (float*)d_out;

  char* ws = (char*)d_ws;
  size_t off = 0;
  auto take = [&](size_t bytes) -> char* {
    char* p = ws + off;
    off = (off + bytes + 255) & ~(size_t)255;
    return p;
  };
  __bf16* wq_p = (__bf16*)take((size_t)D_MODEL * D_MODEL * 2);
  __bf16* wk_p = (__bf16*)take((size_t)D_MODEL * D_MODEL * 2);
  __bf16* wv_p = (__bf16*)take((size_t)D_MODEL * D_MODEL * 2);
  __bf16* wo_p = (__bf16*)take((size_t)D_MODEL * D_MODEL * 2);
  __bf16* w1_p = (__bf16*)take((size_t)D_MODEL * D_FF * 2);
  __bf16* w2_p = (__bf16*)take((size_t)D_FF * D_MODEL * 2);
  __bf16* h    = (__bf16*)take((size_t)TOKENS * D_MODEL * 2);   // LN1 then LN2 output
  float*  x1   = (float*)take((size_t)TOKENS * D_MODEL * 4);    // post-attention residual
  char*   big  = take((size_t)TOKENS * D_FF * 2);               // 32 MB shared region
  __bf16* Qb   = (__bf16*)big;
  __bf16* Kb   = Qb + (size_t)BH * SEQ * D_HEAD;
  __bf16* Vt   = Kb + (size_t)BH * SEQ * D_HEAD;
  __bf16* ctxb = Vt + (size_t)BH * SEQ * D_HEAD;
  __bf16* f1   = (__bf16*)big;   // FFN hidden reuses Q/K/V/ctx after attention

  // 1) pack weights to WMMA B layout (bf16)
  auto pack = [&](const float* W, __bf16* dst, int K, int N) {
    int total = K * N;
    pack_w<<<(total + 255) / 256, 256, 0, stream>>>(W, dst, K, N);
  };
  pack(Wq, wq_p, D_MODEL, D_MODEL);
  pack(Wk, wk_p, D_MODEL, D_MODEL);
  pack(Wv, wv_p, D_MODEL, D_MODEL);
  pack(Wo, wo_p, D_MODEL, D_MODEL);
  pack(W1, w1_p, D_MODEL, D_FF);
  pack(W2, w2_p, D_FF, D_MODEL);

  // 2) h = LN1(x) (bf16)
  ln_kernel<<<TOKENS, 256, 0, stream>>>(x, g1, be1, h);

  // 3) QKV projection (fused, scatter to attention layouts)
  {
    int waves = (TOKENS / 16) * ((3 * D_MODEL) / 64);   // 12288
    gemm_qkv<<<waves / 8, 256, 0, stream>>>(h, wq_p, wk_p, wv_p, bq, bk, bv,
                                            Qb, Kb, Vt);
  }
  // 4) flash attention -> ctx bf16
  attn_kernel<<<BH * (SEQ / 64), 128, 0, stream>>>(Qb, Kb, Vt, mask, ctxb);

  // 5) x1 = x + ctx @ Wo + bo   (fp32)
  {
    int waves = (TOKENS / 16) * (D_MODEL / 64);
    gemm_ws<<<waves / 8, 256, 0, stream>>>(ctxb, wo_p, bo, x, x1, nullptr,
                                           TOKENS, D_MODEL, D_MODEL, 0);
  }
  // 6) h = LN2(x1) (bf16)
  ln_kernel<<<TOKENS, 256, 0, stream>>>(x1, g2, be2, h);

  // 7) f1 = relu(h @ W1 + b1) (bf16)
  {
    int waves = (TOKENS / 16) * (D_FF / 64);
    gemm_ws<<<waves / 8, 256, 0, stream>>>(h, w1_p, b1, nullptr, nullptr, f1,
                                           TOKENS, D_MODEL, D_FF, 1);
  }
  // 8) out = x1 + f1 @ W2 + b2 (fp32)
  {
    int waves = (TOKENS / 16) * (D_MODEL / 64);
    gemm_ws<<<waves / 8, 256, 0, stream>>>(f1, w2_p, b2, x1, out, nullptr,
                                           TOKENS, D_FF, D_MODEL, 0);
  }
}
